// DeepLACForests_90486370992812
// MI455X (gfx1250) — compile-verified, matrix operand order checked by reading, not run on hardware
//
#include <hip/hip_runtime.h>
#include <hip/hip_bf16.h>
#include <stdint.h>

// ---------------------------------------------------------------------------
// DeepLACForests on gfx1250: bf16 WMMA GEMMs + sigmoid tree expansion.
// Memory-bound (268MB output). nodeW is pre-converted once to a padded bf16
// image; the forest GEMM stages both operands into LDS with
// GLOBAL_LOAD_ASYNC_TO_LDS_B64 (ASYNCcnt-tracked, bypasses VGPRs).
// ---------------------------------------------------------------------------

typedef __attribute__((ext_vector_type(16))) __bf16         v16bf;
typedef __attribute__((ext_vector_type(16))) unsigned short v16us;
typedef __attribute__((ext_vector_type(8)))  float          v8f;

#define ENSEMBLE  16
#define CLASS_NUM 10
#define IN_DIM    784
#define HID       128
#define BATCH     8192
#define INTERNAL  255   // 2^(DEPTH-1)-1
#define TOTAL     511   // 2^DEPTH-1

#define LDSTR_A   34    // feats kernel k-stride (17 banks -> conflict-free)
#define ASTR      36    // async-tile k-stride: 72B rows, 8B aligned, no conflicts
#define BCOLS     32    // batch columns per forest block
#define MPAD      256   // 255 internal nodes padded to 256

#if __has_builtin(__builtin_amdgcn_global_load_async_to_lds_b64)
#define USE_ASYNC_LDS 1
#else
#define USE_ASYNC_LDS 0
#endif

#if USE_ASYNC_LDS
// Builtin prototype (from hipcc diagnostic): (v2i AS1*, v2i AS3*, imm, imm)
typedef __attribute__((ext_vector_type(2))) int async_v2i;
typedef __attribute__((address_space(1))) async_v2i* gv2i_p;
typedef __attribute__((address_space(3))) async_v2i* lv2i_p;
typedef __attribute__((address_space(1))) void*      gvoid_p;
typedef __attribute__((address_space(3))) void*      lvoid_p;
#endif

__device__ __forceinline__ unsigned short f2bf(float x) {   // RNE f32->bf16
  unsigned u = __builtin_bit_cast(unsigned, x);
  unsigned r = u + 0x7FFFu + ((u >> 16) & 1u);
  return (unsigned short)(r >> 16);
}
__device__ __forceinline__ float bf2f(unsigned short h) {
  return __builtin_bit_cast(float, (unsigned)h << 16);
}
// K index of low element of VGPR pair p for 16-bit A/B fragments (ISA 7.12.2)
__device__ __forceinline__ int frag_k0(int p, int lane) {
  int kg = (lane >> 4) << 3;                 // lanes 16-31 hold K+8
  return ((p < 4) ? (2 * p) : (16 + 2 * (p - 4))) + kg;
}

// 8-byte global->LDS copy: async (ASYNCcnt) when available, VALU fallback.
__device__ __forceinline__ void g2l_b64(const unsigned short* gsrc, unsigned short* ldst) {
#if USE_ASYNC_LDS
  __builtin_amdgcn_global_load_async_to_lds_b64(
      (gv2i_p)(gvoid_p)(unsigned short*)gsrc,
      (lv2i_p)(lvoid_p)ldst, 0, 0);
#else
  *(unsigned long long*)ldst = *(const unsigned long long*)gsrc;
#endif
}
__device__ __forceinline__ void g2l_join() {
#if USE_ASYNC_LDS
#if __has_builtin(__builtin_amdgcn_s_wait_asynccnt)
  __builtin_amdgcn_s_wait_asynccnt(0);
#else
  asm volatile("s_wait_asynccnt 0" ::: "memory");
#endif
#endif
}

// ---------------------------------------------------------------------------
// Kernel 0: nodeW (16,255,128) f32 -> padded bf16 image (16,256,128).
// 2MB read + 1MB write, once per launch; removes 134M redundant VALU
// conversions from the forest kernel and halves its A-operand L2 traffic.
// ---------------------------------------------------------------------------
__global__ __launch_bounds__(256)
void convert_nodew_kernel(const float* __restrict__ nodeW,
                          unsigned short* __restrict__ nodeWb) {
  int idx = blockIdx.x * 256 + threadIdx.x;   // over 16*256*128 = 524288
  int t = idx >> 15;
  int r = (idx >> 7) & (MPAD - 1);
  int k = idx & (HID - 1);
  float v = (r < INTERNAL) ? nodeW[((size_t)t * INTERNAL + r) * HID + k] : 0.0f;
  nodeWb[idx] = f2bf(v);
}

// ---------------------------------------------------------------------------
// Kernel 1: feats = relu(X @ W1 + b1), stored as bf16 in workspace.
// 128(M) x 128(N) tile per block; 8 waves, wave w owns M-tile w, all 8 N-tiles.
// ---------------------------------------------------------------------------
__global__ __launch_bounds__(256)
void feats_kernel(const float* __restrict__ X, const float* __restrict__ W1,
                  const float* __restrict__ b1, unsigned short* __restrict__ featsB) {
  __shared__ unsigned short ldsX[128][LDSTR_A];   // [m][k] bf16
  __shared__ unsigned short ldsW[HID][LDSTR_A];   // [n][k] bf16 (W1 transposed)
  const int tid  = threadIdx.x;
  const int lane = tid & 31;
  const int w    = tid >> 5;
  const int m0   = blockIdx.x * 128;

  v8f acc[8];
  for (int nt = 0; nt < 8; ++nt)
    for (int r = 0; r < 8; ++r) acc[nt][r] = 0.0f;

  for (int k0 = 0; k0 < IN_DIM; k0 += 32) {
    __syncthreads();
    { // stage X chunk: 128 rows x 32 k, zero-pad K tail (784 = 24*32+16)
      int m  = tid >> 1;
      int kb = (tid & 1) * 16;
      const float* src = X + (size_t)(m0 + m) * IN_DIM + k0 + kb;
      if (k0 + 32 < IN_DIM)
        __builtin_prefetch(src + 32, 0, 1);     // global_prefetch next chunk
#pragma unroll
      for (int j = 0; j < 16; ++j) {
        int k = kb + j;
        float v = (k0 + k < IN_DIM) ? src[j] : 0.0f;
        ldsX[m][k] = f2bf(v);
      }
    }
    { // stage W1 chunk transposed: [n][k]
      int k  = tid >> 3;
      int nb = (tid & 7) * 16;
      bool ok = (k0 + k) < IN_DIM;
      const float* src = W1 + (size_t)(k0 + k) * HID + nb;
#pragma unroll
      for (int j = 0; j < 16; ++j)
        ldsW[nb + j][k] = ok ? f2bf(src[j]) : (unsigned short)0;
    }
    __syncthreads();

    v16us au;
    int arow = w * 16 + (lane & 15);
#pragma unroll
    for (int p = 0; p < 8; ++p) {
      int kk = frag_k0(p, lane);
      au[2 * p]     = ldsX[arow][kk];
      au[2 * p + 1] = ldsX[arow][kk + 1];
    }
    v16bf afr = __builtin_bit_cast(v16bf, au);
#pragma unroll
    for (int nt = 0; nt < 8; ++nt) {
      v16us bu;
      int bcol = nt * 16 + (lane & 15);
#pragma unroll
      for (int p = 0; p < 8; ++p) {
        int kk = frag_k0(p, lane);
        bu[2 * p]     = ldsW[bcol][kk];
        bu[2 * p + 1] = ldsW[bcol][kk + 1];
      }
      v16bf bfr = __builtin_bit_cast(v16bf, bu);
      acc[nt] = __builtin_amdgcn_wmma_f32_16x16x32_bf16(
          false, afr, false, bfr, (short)0, acc[nt], false, false);
    }
  }

  // epilogue: bias + relu, store bf16 feats
#pragma unroll
  for (int nt = 0; nt < 8; ++nt) {
    int n = nt * 16 + (lane & 15);
    float bias = b1[n];
#pragma unroll
    for (int r = 0; r < 8; ++r) {
      int m = m0 + w * 16 + r + ((lane >> 4) << 3);
      float v = acc[nt][r] + bias;
      v = v > 0.0f ? v : 0.0f;
      featsB[(size_t)m * HID + n] = f2bf(v);
    }
  }
}

// ---------------------------------------------------------------------------
// Kernel 2: prediction = feats @ W2 + b2  (tiny; memory-trivial)
// ---------------------------------------------------------------------------
__global__ __launch_bounds__(256)
void pred_kernel(const unsigned short* __restrict__ featsB,
                 const float* __restrict__ W2, const float* __restrict__ b2,
                 float* __restrict__ out) {
  __shared__ float w2s[HID][CLASS_NUM];
  __shared__ float b2s[CLASS_NUM];
  int tid = threadIdx.x;
  for (int i = tid; i < HID * CLASS_NUM; i += 256)
    w2s[i / CLASS_NUM][i % CLASS_NUM] = W2[i];
  if (tid < CLASS_NUM) b2s[tid] = b2[tid];
  __syncthreads();

  int b = blockIdx.x * 256 + tid;
  float acc[CLASS_NUM];
#pragma unroll
  for (int c = 0; c < CLASS_NUM; ++c) acc[c] = b2s[c];
  const unsigned short* row = featsB + (size_t)b * HID;
  for (int d = 0; d < HID; ++d) {
    float f = bf2f(row[d]);
#pragma unroll
    for (int c = 0; c < CLASS_NUM; ++c) acc[c] += f * w2s[d][c];
  }
#pragma unroll
  for (int c = 0; c < CLASS_NUM; ++c)
    out[(size_t)b * CLASS_NUM + c] = acc[c];
}

// ---------------------------------------------------------------------------
// Kernel 3: per (tree, 32-batch-column) tile:
//   dec = sigmoid(nodeWb[t] @ feats^T + nodeb[t])  via WMMA (M=256,K=128),
//   operands staged bf16->LDS with async global->LDS copies,
//   then heap path products: child 2m <- parent*d, child 2m+1 <- parent*(1-d)
// ---------------------------------------------------------------------------
__global__ __launch_bounds__(256)
void forest_kernel(const unsigned short* __restrict__ featsB,
                   const unsigned short* __restrict__ nodeWb,
                   const float* __restrict__ nodeb,
                   float* __restrict__ out) {
  __shared__ unsigned short ldsA[MPAD][ASTR];    // nodeW tile [m][k] bf16
  __shared__ unsigned short ldsB[BCOLS][ASTR];   // feats^T tile [b][k] bf16
  __shared__ float dec[MPAD][BCOLS + 1];         // sigmoid decisions
  const int tid  = threadIdx.x;
  const int lane = tid & 31;
  const int w    = tid >> 5;
  const int b0   = blockIdx.x * BCOLS;
  const int t    = blockIdx.y;
  const unsigned short* nwb = nodeWb + (size_t)t * MPAD * HID;

  v8f acc[2][2];
  for (int mi = 0; mi < 2; ++mi)
    for (int ni = 0; ni < 2; ++ni)
      for (int r = 0; r < 8; ++r) acc[mi][ni][r] = 0.0f;

  for (int k0 = 0; k0 < HID; k0 += 32) {
    __syncthreads();
    // stage nodeW tile: 256 rows x 32 k bf16 = 2048 b64 segments (8/thread)
#pragma unroll
    for (int i = 0; i < 8; ++i) {
      int s   = tid + 256 * i;          // 0..2047
      int row = s >> 3;
      int seg = (s & 7) * 4;            // ushort offset of 8B segment
      g2l_b64(nwb + (size_t)row * HID + k0 + seg, &ldsA[row][seg]);
    }
    // stage feats^T tile: 32 rows x 32 k bf16 = 256 b64 segments (1/thread)
    {
      int row = tid >> 3;
      int seg = (tid & 7) * 4;
      g2l_b64(featsB + (size_t)(b0 + row) * HID + k0 + seg, &ldsB[row][seg]);
    }
    g2l_join();
    __syncthreads();

#pragma unroll
    for (int ni = 0; ni < 2; ++ni) {
      v16us bu;
      int bcol = ni * 16 + (lane & 15);
#pragma unroll
      for (int p = 0; p < 8; ++p) {
        int kk = frag_k0(p, lane);
        bu[2 * p]     = ldsB[bcol][kk];
        bu[2 * p + 1] = ldsB[bcol][kk + 1];
      }
      v16bf bfr = __builtin_bit_cast(v16bf, bu);
#pragma unroll
      for (int mi = 0; mi < 2; ++mi) {
        v16us au;
        int arow = (2 * w + mi) * 16 + (lane & 15);
#pragma unroll
        for (int p = 0; p < 8; ++p) {
          int kk = frag_k0(p, lane);
          au[2 * p]     = ldsA[arow][kk];
          au[2 * p + 1] = ldsA[arow][kk + 1];
        }
        v16bf afr = __builtin_bit_cast(v16bf, au);
        acc[mi][ni] = __builtin_amdgcn_wmma_f32_16x16x32_bf16(
            false, afr, false, bfr, (short)0, acc[mi][ni], false, false);
      }
    }
  }

  // sigmoid + bias -> dec LDS (C-fragment layout: M = r + 8*(lane>>4))
  const float* nbp = nodeb + (size_t)t * INTERNAL;
#pragma unroll
  for (int mi = 0; mi < 2; ++mi)
#pragma unroll
    for (int ni = 0; ni < 2; ++ni) {
      int bcol = ni * 16 + (lane & 15);
#pragma unroll
      for (int r = 0; r < 8; ++r) {
        int m = (2 * w + mi) * 16 + r + ((lane >> 4) << 3);
        float x = acc[mi][ni][r] + ((m < INTERNAL) ? nbp[m] : 0.0f);
        dec[m][bcol] = 1.0f / (1.0f + __expf(-x));
      }
    }
  __syncthreads();

  // tree expansion: heap index h in [1,511]; factor at ancestor a = h>>k:
  // even child takes d[a], odd child takes 1-d[a]. Writes coalesced over b.
  float* dst = out + (size_t)BATCH * CLASS_NUM + (size_t)t * TOTAL * BATCH + b0;
  int b = tid & (BCOLS - 1);
  for (int h = 1 + (tid >> 5); h <= TOTAL; h += 8) {
    float r = 1.0f;
    unsigned m = (unsigned)h;
    while (m > 1) {
      float d = dec[(m >> 1) - 1][b];
      r *= (m & 1) ? (1.0f - d) : d;
      m >>= 1;
    }
    dst[(size_t)(h - 1) * BATCH + b] = r;
  }
}

// ---------------------------------------------------------------------------
extern "C" void kernel_launch(void* const* d_in, const int* in_sizes, int n_in,
                              void* d_out, int out_size, void* d_ws, size_t ws_size,
                              hipStream_t stream) {
  const float* X     = (const float*)d_in[0];
  const float* W1    = (const float*)d_in[1];
  const float* b1    = (const float*)d_in[2];
  const float* W2    = (const float*)d_in[3];
  const float* b2    = (const float*)d_in[4];
  const float* nodeW = (const float*)d_in[5];
  const float* nodeb = (const float*)d_in[6];
  float* out = (float*)d_out;

  unsigned short* featsB = (unsigned short*)d_ws;        // 8192*128 bf16 = 2 MB
  unsigned short* nodeWb = featsB + (size_t)BATCH * HID; // 16*256*128 bf16 = 1 MB

  convert_nodew_kernel<<<(ENSEMBLE * MPAD * HID) / 256, 256, 0, stream>>>(nodeW, nodeWb);
  feats_kernel<<<BATCH / 128, 256, 0, stream>>>(X, W1, b1, featsB);
  pred_kernel<<<BATCH / 256, 256, 0, stream>>>(featsB, W2, b2, out);
  forest_kernel<<<dim3(BATCH / BCOLS, ENSEMBLE), 256, 0, stream>>>(
      featsB, nodeWb, nodeb, out);
}
